// DynamicEdgeConvPN_6038724018830
// MI455X (gfx1250) — compile-verified, matrix operand order checked by reading, not run on hardware
//
#include <hip/hip_runtime.h>
#include <hip/hip_bf16.h>

// Problem constants (from the reference): B=32, P=1024, K=16, C=64, H=128
#define NPTS   32768      // B*P
#define CC     64
#define HH     128
#define PP     1024
#define BB     32
#define KNN    16
#define NCOL   384        // T(128) | U(128) | S(128) fused output columns

typedef __attribute__((ext_vector_type(16))) __bf16 v16bf;
typedef __attribute__((ext_vector_type(8)))  float  v8f;

// ---------------------------------------------------------------------------
// Workspace layout (bytes, 256-aligned):
//   [0)            xb16  : NPTS*CC   bf16  = 4,194,304
//   [4,194,304)    WcatT : NCOL*CC   bf16  =    49,152
//   [4,243,456)    knn   : NPTS*KNN  int   = 2,097,152
//   [6,340,608)    TUS   : NPTS*NCOL f32   = 50,331,648
// total = 56,672,256 bytes
// ---------------------------------------------------------------------------
#define WS_OFF_WCAT 4194304
#define WS_OFF_KNN  4243456
#define WS_OFF_TUS  6340608

// ---- Kernel 1: convert x (f32) -> bf16, row-major 64-stride ----------------
__global__ void cvt_x_kernel(const float* __restrict__ x, __bf16* __restrict__ xb, int n) {
    int i = blockIdx.x * blockDim.x + threadIdx.x;
    if (i < n) xb[i] = (__bf16)x[i];
}

// ---- Kernel 2: build fused transposed weights WcatT[n][k] (bf16) ----------
//   n <  128 : (W_e[k][n] - W_e[64+k][n])        -> T columns
//   n <  256 :  W_e[64+k][n-128]                 -> U columns
//   n <  384 :  W_sc[k][n-256]                   -> S columns
__global__ void build_w_kernel(const float* __restrict__ W_e,
                               const float* __restrict__ W_sc,
                               __bf16* __restrict__ WcatT) {
    int i = blockIdx.x * blockDim.x + threadIdx.x;
    if (i >= NCOL * CC) return;
    int n = i / CC, k = i % CC;
    float v;
    if (n < HH)            v = W_e[k * HH + n] - W_e[(CC + k) * HH + n];
    else if (n < 2 * HH)   v = W_e[(CC + k) * HH + (n - HH)];
    else                   v = W_sc[k * HH + (n - 2 * HH)];
    WcatT[n * CC + k] = (__bf16)v;
}

// ---- Kernel 3: per-batch KNN (K=16 smallest d2, self excluded) ------------
// One block per batch, 1024 threads (one point each). Positions staged in LDS.
// 16-deep register insertion network keeps lanes converged.
__global__ void __launch_bounds__(PP) knn_kernel(const float* __restrict__ pos,
                                                 int* __restrict__ knn_idx) {
    __shared__ float sx[PP], sy[PP], sz[PP];
    const int b = blockIdx.x;
    const int p = threadIdx.x;
    const int g = b * PP + p;
    const float px = pos[g * 3 + 0];
    const float py = pos[g * 3 + 1];
    const float pz = pos[g * 3 + 2];
    sx[p] = px; sy[p] = py; sz[p] = pz;
    __syncthreads();

    float dist[KNN];
    int   idx[KNN];
#pragma unroll
    for (int t = 0; t < KNN; ++t) { dist[t] = 3.0e38f; idx[t] = 0; }

    for (int j = 0; j < PP; ++j) {
        float dx = sx[j] - px, dy = sy[j] - py, dz = sz[j] - pz;
        float d = dx * dx + dy * dy + dz * dz;
        d = (j == p) ? 3.4e38f : d;           // exclude self (matches eye*max/2)
        if (d < dist[KNN - 1]) {
            int id = j;
#pragma unroll
            for (int t = 0; t < KNN; ++t) {   // sorted insert, drop largest
                bool  sw = d < dist[t];
                float td = sw ? dist[t] : d;
                int   ti = sw ? idx[t]  : id;
                dist[t]  = sw ? d  : dist[t];
                idx[t]   = sw ? id : idx[t];
                d = td; id = ti;
            }
        }
    }
#pragma unroll
    for (int t = 0; t < KNN; ++t)
        knn_idx[g * KNN + t] = b * PP + idx[t];   // store GLOBAL neighbor index
}

// ---- Kernel 4: fused GEMM  TUS = xb16(32768x64) @ WcatT^T(64x384) ---------
// One wave owns a 16-row tile and sweeps 6 column tiles (block of 4 waves
// covers all 24). A fragments are loaded ONCE per wave (A traffic 16 MB total
// instead of 96 MB); 12 v_wmma_f32_16x16x32_bf16 per wave with independent
// accumulators, so WMMAs pipeline without D->A/B hazards.
__global__ void __launch_bounds__(128) gemm_kernel(const __bf16* __restrict__ xb,
                                                   const __bf16* __restrict__ wT,
                                                   float* __restrict__ TUS) {
    const int lane = threadIdx.x & 31;
    const int wave = threadIdx.x >> 5;
    const int rt   = blockIdx.x;                 // row tile 0..2047
    const int m    = lane & 15;
    const int kh   = (lane >> 4) * 8;            // K-half select: 0 or 8

    union frag { uint4 u[2]; v16bf v; };

    // ISA 16-bit A 16x32 layout: lane half kh -> K = kh+0..7 and 16+kh+0..7
    const __bf16* arow = xb + (size_t)(rt * 16 + m) * CC;  // 16B-aligned
    frag a0, a1;
    a0.u[0] = *(const uint4*)(arow + kh);
    a0.u[1] = *(const uint4*)(arow + 16 + kh);
    a1.u[0] = *(const uint4*)(arow + 32 + kh);
    a1.u[1] = *(const uint4*)(arow + 48 + kh);

    float* outrow = TUS + (size_t)(rt * 16) * NCOL + m;

#pragma unroll
    for (int t = 0; t < 6; ++t) {
        const int ct = wave * 6 + t;             // col tile 0..23
        const __bf16* brow = wT + (size_t)(ct * 16 + m) * CC;  // pre-transposed
        frag b0, b1;
        b0.u[0] = *(const uint4*)(brow + kh);
        b0.u[1] = *(const uint4*)(brow + 16 + kh);
        b1.u[0] = *(const uint4*)(brow + 32 + kh);
        b1.u[1] = *(const uint4*)(brow + 48 + kh);

        v8f c = {0.f, 0.f, 0.f, 0.f, 0.f, 0.f, 0.f, 0.f};
        c = __builtin_amdgcn_wmma_f32_16x16x32_bf16(false, a0.v, false, b0.v,
                                                    (short)0, c, false, false);
        c = __builtin_amdgcn_wmma_f32_16x16x32_bf16(false, a1.v, false, b1.v,
                                                    (short)0, c, false, false);

        // C/D layout: VGPR r, lane L -> M = r + 8*(L>=16), N = L&15
        float* outp = outrow + ct * 16;
#pragma unroll
        for (int r = 0; r < 8; ++r)
            outp[(size_t)(r + kh) * NCOL] = c[r];
    }
}

// ---- Kernel 5: gather + relu + max over K + shortcut ----------------------
// out[p][h] = max(0, max_k (T[p][h] + U[idx[p][k]][h] + b_e[h])) + S[p][h] + b_sc[h]
// 128 threads per point (h = channel); U gathers are 512B-contiguous per row,
// per-batch U slice is 512KB -> L2 resident.
__global__ void __launch_bounds__(256) fuse_kernel(const float* __restrict__ TUS,
                                                   const int* __restrict__ knn_idx,
                                                   const float* __restrict__ b_e,
                                                   const float* __restrict__ b_sc,
                                                   float* __restrict__ out) {
    const int p = blockIdx.x * 2 + (threadIdx.x >> 7);
    const int h = threadIdx.x & (HH - 1);
    const float t = TUS[(size_t)p * NCOL + h] + b_e[h];
    const int* ki = knn_idx + (size_t)p * KNN;
    float acc = 0.0f;                       // max_k relu(v) == max(0, max_k v)
#pragma unroll
    for (int k = 0; k < KNN; ++k) {
        int j = ki[k];
        acc = fmaxf(acc, t + TUS[(size_t)j * NCOL + HH + h]);
    }
    out[(size_t)p * HH + h] = acc + TUS[(size_t)p * NCOL + 2 * HH + h] + b_sc[h];
}

// ---------------------------------------------------------------------------
extern "C" void kernel_launch(void* const* d_in, const int* in_sizes, int n_in,
                              void* d_out, int out_size, void* d_ws, size_t ws_size,
                              hipStream_t stream) {
    const float* x    = (const float*)d_in[0];   // (32768, 64)
    const float* pos  = (const float*)d_in[1];   // (32768, 3)
    // d_in[2] = batch (int64) — implied by p/PP, unused
    const float* W_e  = (const float*)d_in[3];   // (128, 128)
    const float* b_e  = (const float*)d_in[4];   // (128,)
    const float* W_sc = (const float*)d_in[5];   // (64, 128)
    const float* b_sc = (const float*)d_in[6];   // (128,)
    float* out = (float*)d_out;                  // (32768, 128)

    char* ws = (char*)d_ws;
    __bf16* xb16  = (__bf16*)ws;
    __bf16* WcatT = (__bf16*)(ws + WS_OFF_WCAT);
    int*    knn   = (int*)(ws + WS_OFF_KNN);
    float*  TUS   = (float*)(ws + WS_OFF_TUS);

    (void)in_sizes; (void)n_in; (void)out_size; (void)ws_size;

    // 1) x -> bf16
    {
        int n = NPTS * CC;
        cvt_x_kernel<<<(n + 255) / 256, 256, 0, stream>>>(x, xb16, n);
    }
    // 2) fused transposed weights
    build_w_kernel<<<(NCOL * CC + 255) / 256, 256, 0, stream>>>(W_e, W_sc, WcatT);
    // 3) KNN (one block per batch)
    knn_kernel<<<BB, PP, 0, stream>>>(pos, knn);
    // 4) WMMA GEMM: TUS = xb16 @ Wcat (one wave per 16-row tile, sweeps cols)
    gemm_kernel<<<NPTS / 16, 128, 0, stream>>>(xb16, WcatT, TUS);
    // 5) gather + relu + max + shortcut
    fuse_kernel<<<NPTS / 2, 256, 0, stream>>>(TUS, knn, b_e, b_sc, out);
}
